// TAttention_29583734735457
// MI455X (gfx1250) — matrix operand order repaired and checked by reading, not compiled, not run on hardware
//
#include <hip/hip_runtime.h>

// Problem constants (B=32, S=1024, D=256, H=8, HD=32)
#define TOK   32768   // B*S
#define DD    256
#define SS    1024
#define HH    8
#define HDD   32
#define EPSLN 1e-5f

typedef __attribute__((ext_vector_type(16))) __bf16 v16bf;
typedef __attribute__((ext_vector_type(8)))  __bf16 v8bf;
typedef __attribute__((ext_vector_type(8)))  float  v8f;

static __device__ __forceinline__ v8f v8f_zero() {
  v8f z = {0.f,0.f,0.f,0.f,0.f,0.f,0.f,0.f};
  return z;
}
static __device__ __forceinline__ v16bf join16(v8bf lo, v8bf hi) {
  return __builtin_shufflevector(lo, hi, 0,1,2,3,4,5,6,7,8,9,10,11,12,13,14,15);
}

// DPP8 selector: lane i reads lane sel[i] within its 8-lane group
#define DPP8SEL(a,b,c,d,e,f,g,h) \
  ((a)|((b)<<3)|((c)<<6)|((d)<<9)|((e)<<12)|((f)<<15)|((g)<<18)|((h)<<21))
#define DPP8_XOR1 DPP8SEL(1,0,3,2,5,4,7,6)
#define DPP8_XOR2 DPP8SEL(2,3,0,1,6,7,4,5)
#define DPP8_XOR4 DPP8SEL(4,5,6,7,0,1,2,3)
#define DPP16_ROW_MIRROR 0x140

// Max-reduce across the 16 lanes of each wave half; pure VALU (DPP), no LDS.
static __device__ __forceinline__ float dpp_max16(float x) {
  x = fmaxf(x, __int_as_float(__builtin_amdgcn_mov_dpp8(__float_as_int(x), DPP8_XOR1)));
  x = fmaxf(x, __int_as_float(__builtin_amdgcn_mov_dpp8(__float_as_int(x), DPP8_XOR2)));
  x = fmaxf(x, __int_as_float(__builtin_amdgcn_mov_dpp8(__float_as_int(x), DPP8_XOR4)));
  x = fmaxf(x, __int_as_float(__builtin_amdgcn_mov_dpp(__float_as_int(x),
                                                       DPP16_ROW_MIRROR, 0xf, 0xf, true)));
  return x;
}

// ---------------------------------------------------------------------------
// Weight convert + transpose: Wt[n][k] = bf16(W[k][n]),  256x256
// ---------------------------------------------------------------------------
__global__ void wconvT_kernel(const float* __restrict__ W, __bf16* __restrict__ Wt) {
  int idx = blockIdx.x * 256 + threadIdx.x;          // 0 .. 65535
  int k = idx >> 8;
  int n = idx & 255;
  Wt[(size_t)n * DD + k] = (__bf16)W[(size_t)k * DD + n];
}

// ---------------------------------------------------------------------------
// LayerNorm: one wave per row (32 lanes x 8 floats = 256).
// HASRES: input = X + R (residual).  OUTF: also store f32 result.
// ---------------------------------------------------------------------------
template<bool HASRES, bool OUTF>
__global__ void layernorm_kernel(const float* __restrict__ X,
                                 const float* __restrict__ R,
                                 const float* __restrict__ gamma,
                                 const float* __restrict__ beta,
                                 __bf16* __restrict__ outb,
                                 float* __restrict__ outf) {
  const int lane = threadIdx.x & 31;
  const int wave = threadIdx.x >> 5;
  const int row  = blockIdx.x * 8 + wave;
  const size_t base = (size_t)row * DD + lane * 8;

  const float4* xp = (const float4*)(X + base);
  float4 a0 = xp[0], a1 = xp[1];
  float v[8] = {a0.x, a0.y, a0.z, a0.w, a1.x, a1.y, a1.z, a1.w};
  if constexpr (HASRES) {
    const float4* rp = (const float4*)(R + base);
    float4 r0 = rp[0], r1 = rp[1];
    v[0]+=r0.x; v[1]+=r0.y; v[2]+=r0.z; v[3]+=r0.w;
    v[4]+=r1.x; v[5]+=r1.y; v[6]+=r1.z; v[7]+=r1.w;
  }
  float s = 0.f;
#pragma unroll
  for (int i = 0; i < 8; ++i) s += v[i];
#pragma unroll
  for (int off = 1; off < 32; off <<= 1) s += __shfl_xor(s, off, 32);
  const float mu = s * (1.f / DD);
  float var = 0.f;
#pragma unroll
  for (int i = 0; i < 8; ++i) { float d = v[i] - mu; var += d * d; }
#pragma unroll
  for (int off = 1; off < 32; off <<= 1) var += __shfl_xor(var, off, 32);
  var *= (1.f / DD);
  const float rstd = rsqrtf(var + EPSLN);

  const float4* gp = (const float4*)(gamma + lane * 8);
  const float4* bp = (const float4*)(beta  + lane * 8);
  float4 g0 = gp[0], g1 = gp[1], b0 = bp[0], b1 = bp[1];
  float gv[8] = {g0.x,g0.y,g0.z,g0.w,g1.x,g1.y,g1.z,g1.w};
  float bv[8] = {b0.x,b0.y,b0.z,b0.w,b1.x,b1.y,b1.z,b1.w};

  float y[8];
  v8bf ob;
#pragma unroll
  for (int i = 0; i < 8; ++i) {
    y[i] = (v[i] - mu) * rstd * gv[i] + bv[i];
    ob[i] = (__bf16)y[i];
  }
  *(v8bf*)(outb + base) = ob;
  if constexpr (OUTF) {
    float4* op = (float4*)(outf + base);
    op[0] = make_float4(y[0], y[1], y[2], y[3]);
    op[1] = make_float4(y[4], y[5], y[6], y[7]);
  }
}

// ---------------------------------------------------------------------------
// WMMA GEMM:  C[M=32768][N=256] = A[M][K=256] (bf16, row major)
//                               x Bt[N][K]    (bf16, pre-transposed weight)
// Block = 256 thr (8 waves); wave computes 16 rows x 64 cols (4 WMMA tiles).
// Epilogue:
//   EPI 0: store bf16 row-major
//   EPI 1: store bf16 transposed per head: out[(b*H+h)*HD+hd][s]  (for V)
//   EPI 2: +bias, ReLU, store bf16 row-major
//   EPI 3: outf[m][n] += acc + bias[n]   (f32 residual accumulate)
// ---------------------------------------------------------------------------
template<int EPI>
__global__ void gemm_wmma_kernel(const __bf16* __restrict__ A,
                                 const __bf16* __restrict__ Bt,
                                 const float* __restrict__ bias,
                                 __bf16* __restrict__ outb,
                                 float* __restrict__ outf) {
  const int lane = threadIdx.x & 31;
  const int wave = threadIdx.x >> 5;
  const int half = lane >> 4;      // 0/1
  const int l16  = lane & 15;
  const int kb   = half * 16;

  const int m0 = blockIdx.y * 128 + wave * 16;
  const int n0 = blockIdx.x * 64;
  const int arow = m0 + l16;

  v8f acc[4];
#pragma unroll
  for (int t = 0; t < 4; ++t) acc[t] = v8f_zero();

  const __bf16* arp = A + (size_t)arow * DD;
#pragma unroll
  for (int k0 = 0; k0 < DD; k0 += 32) {
    if (k0 + 32 < DD) __builtin_prefetch(arp + k0 + 32, 0, 1);  // global_prefetch_b8
    v8bf alo = *(const v8bf*)(arp + k0 + kb);
    v8bf ahi = *(const v8bf*)(arp + k0 + kb + 8);
    v16bf afrag = join16(alo, ahi);
#pragma unroll
    for (int nt = 0; nt < 4; ++nt) {
      const __bf16* bp = Bt + (size_t)(n0 + nt * 16 + l16) * DD + k0 + half * 16;
      v16bf bfrag = *(const v16bf*)bp;
      acc[nt] = __builtin_amdgcn_wmma_f32_16x16x32_bf16(
          false, afrag, false, bfrag, (short)0, acc[nt], false, false);
    }
  }

#pragma unroll
  for (int nt = 0; nt < 4; ++nt) {
#pragma unroll
    for (int r = 0; r < 8; ++r) {
      const int m = m0 + half * 8 + r;
      const int n = n0 + nt * 16 + l16;
      float vv = acc[nt][r];
      if constexpr (EPI == 2) {
        vv += bias[n];
        vv = vv > 0.f ? vv : 0.f;
      }
      if constexpr (EPI == 0 || EPI == 2) {
        outb[(size_t)m * DD + n] = (__bf16)vv;
      } else if constexpr (EPI == 1) {
        const int b  = m >> 10;        // / S
        const int sx = m & (SS - 1);
        const int h  = n >> 5;         // / HD
        const int hd = n & (HDD - 1);
        outb[(((size_t)(b * HH + h) * HDD + hd) << 10) + sx] = (__bf16)vv;
      } else { // EPI == 3
        const size_t idx = (size_t)m * DD + n;
        outf[idx] = outf[idx] + vv + bias[n];
      }
    }
  }
}

// ---------------------------------------------------------------------------
// Flash attention (no 1/sqrt(d) scale per reference).
// Block = 128 thr (4 waves); each wave owns one (b, h, 16-query-row) tile and
// streams K in chunks of 32 keys: 2 score WMMAs + DPP tile-max softmax +
// 3 accumulate WMMAs (ctx lo, ctx hi, row-sum via all-ones B).
// Q/K row-major bf16 [tok][D]; V transposed bf16 [(b*H+h)*HD+hd][s].
// ---------------------------------------------------------------------------
__global__ void attn_flash_kernel(const __bf16* __restrict__ Q,
                                  const __bf16* __restrict__ Km,
                                  const __bf16* __restrict__ Vt,
                                  float* __restrict__ ctx) {
  __shared__ __bf16 Pl[4][16 * 32];
  const int lane = threadIdx.x & 31;
  const int wave = threadIdx.x >> 5;
  const int half = lane >> 4;
  const int l16  = lane & 15;
  const int kb   = half * 16;

  const int tile = blockIdx.x * 4 + wave;   // b*H*64 + h*64 + qt
  const int qt = tile & 63;
  const int h  = (tile >> 6) & (HH - 1);
  const int b  = tile >> 9;
  const int q0 = qt * 16;

  // Q A-fragment (persistent): rows q0+l16, K = head dim 0..31
  const __bf16* qp = Q + ((size_t)b * SS + q0 + l16) * DD + h * HDD;
  v16bf qfrag = join16(*(const v8bf*)(qp + kb), *(const v8bf*)(qp + kb + 8));

  // all-ones B matrix: turns the 3rd accumulate WMMA into a row-sum
  v16bf ones;
#pragma unroll
  for (int i = 0; i < 16; ++i) ones[i] = (__bf16)1.0f;

  float mcur = -3.0e38f;                 // running max (shared by this half's rows)
  v8f acc0 = v8f_zero(), acc1 = v8f_zero(), accl = v8f_zero();

  __bf16* pl = &Pl[wave][0];
  const __bf16* kbase = Km + (size_t)b * SS * DD + h * HDD + half * 16;
  const __bf16* vbase = Vt + (((size_t)(b * HH + h) * HDD + l16) << 10) + half * 16;

  for (int kk0 = 0; kk0 < SS; kk0 += 32) {
    // ---- scores: two 16x16 tiles (keys kk0..+15, kk0+16..+31), K=HD=32
    const __bf16* kp0 = kbase + (size_t)(kk0 + l16) * DD;
    v16bf kf0 = *(const v16bf*)kp0;
    v16bf kf1 = *(const v16bf*)(kp0 + (size_t)16 * DD);
    v8f s0 = __builtin_amdgcn_wmma_f32_16x16x32_bf16(false, qfrag, false, kf0,
                                                     (short)0, v8f_zero(), false, false);
    v8f s1 = __builtin_amdgcn_wmma_f32_16x16x32_bf16(false, qfrag, false, kf1,
                                                     (short)0, v8f_zero(), false, false);

    // ---- tile max (upper bound of per-row max; exact after normalization)
    float tm = fmaxf(s0[0], s1[0]);
#pragma unroll
    for (int r = 1; r < 8; ++r) tm = fmaxf(tm, fmaxf(s0[r], s1[r]));
    tm = dpp_max16(tm);                       // VALU-only 16-lane reduce
    const float mnew  = fmaxf(mcur, tm);
    const float alpha = __expf(mcur - mnew);
    mcur = mnew;

    // ---- P = exp(s - mnew), staged to LDS for C->A relayout
#pragma unroll
    for (int r = 0; r < 8; ++r) {
      float p0 = __expf(s0[r] - mnew);
      float p1 = __expf(s1[r] - mnew);
      pl[(half * 8 + r) * 32 + l16]      = (__bf16)p0;
      pl[(half * 8 + r) * 32 + 16 + l16] = (__bf16)p1;
    }
#pragma unroll
    for (int r = 0; r < 8; ++r) {
      acc0[r] *= alpha;
      acc1[r] *= alpha;
      accl[r] *= alpha;
    }
    asm volatile("s_wait_dscnt 0" ::: "memory");
    v16bf pfrag = join16(*(const v8bf*)(pl + l16 * 32 + kb),
                         *(const v8bf*)(pl + l16 * 32 + kb + 8));

    // ---- accumulate: ctx (hd 0-15, 16-31) and row-sum l (B = ones)
    const __bf16* vp0 = vbase + kk0;
    v16bf vf0 = *(const v16bf*)vp0;
    v16bf vf1 = *(const v16bf*)(vp0 + ((size_t)16 << 10));
    acc0 = __builtin_amdgcn_wmma_f32_16x16x32_bf16(false, pfrag, false, vf0,
                                                   (short)0, acc0, false, false);
    acc1 = __builtin_amdgcn_wmma_f32_16x16x32_bf16(false, pfrag, false, vf1,
                                                   (short)0, acc1, false, false);
    accl = __builtin_amdgcn_wmma_f32_16x16x32_bf16(false, pfrag, false, ones,
                                                   (short)0, accl, false, false);
  }

  // ---- normalize and store ctx (f32, row-major [tok][D])
#pragma unroll
  for (int r = 0; r < 8; ++r) {
    const int m = q0 + half * 8 + r;
    const float inv = 1.f / accl[r];   // every column copy holds its row's sum
    float* cp = ctx + ((size_t)b * SS + m) * DD + h * HDD;
    cp[l16]      = acc0[r] * inv;
    cp[16 + l16] = acc1[r] * inv;
  }
}

// ---------------------------------------------------------------------------
// Launch
// ---------------------------------------------------------------------------
extern "C" void kernel_launch(void* const* d_in, const int* in_sizes, int n_in,
                              void* d_out, int out_size, void* d_ws, size_t ws_size,
                              hipStream_t stream) {
  const float* x   = (const float*)d_in[0];
  const float* wq  = (const float*)d_in[1];
  const float* wk  = (const float*)d_in[2];
  const float* wv  = (const float*)d_in[3];
  const float* g1  = (const float*)d_in[4];
  const float* be1 = (const float*)d_in[5];
  const float* g2  = (const float*)d_in[6];
  const float* be2 = (const float*)d_in[7];
  const float* wf1 = (const float*)d_in[8];
  const float* bf1 = (const float*)d_in[9];
  const float* wf2 = (const float*)d_in[10];
  const float* bf2 = (const float*)d_in[11];
  float* out = (float*)d_out;

  char* ws = (char*)d_ws;
  size_t off = 0;
  auto alloc = [&](size_t bytes) -> void* {
    void* p = ws + off;
    off += (bytes + 255) & ~(size_t)255;
    return p;
  };
  __bf16* xn   = (__bf16*)alloc((size_t)TOK * DD * 2);
  __bf16* qb   = (__bf16*)alloc((size_t)TOK * DD * 2);
  __bf16* kbuf = (__bf16*)alloc((size_t)TOK * DD * 2);
  __bf16* vT   = (__bf16*)alloc((size_t)TOK * DD * 2);
  __bf16* ab   = (__bf16*)alloc((size_t)TOK * DD * 2);
  __bf16* hid  = (__bf16*)alloc((size_t)TOK * DD * 2);
  float*  ctx  = (float*) alloc((size_t)TOK * DD * 4);
  __bf16* wqT  = (__bf16*)alloc((size_t)DD * DD * 2);
  __bf16* wkT  = (__bf16*)alloc((size_t)DD * DD * 2);
  __bf16* wvT  = (__bf16*)alloc((size_t)DD * DD * 2);
  __bf16* wf1T = (__bf16*)alloc((size_t)DD * DD * 2);
  __bf16* wf2T = (__bf16*)alloc((size_t)DD * DD * 2);

  // 1) convert + transpose weights to bf16
  wconvT_kernel<<<256, 256, 0, stream>>>(wq,  wqT);
  wconvT_kernel<<<256, 256, 0, stream>>>(wk,  wkT);
  wconvT_kernel<<<256, 256, 0, stream>>>(wv,  wvT);
  wconvT_kernel<<<256, 256, 0, stream>>>(wf1, wf1T);
  wconvT_kernel<<<256, 256, 0, stream>>>(wf2, wf2T);

  // 2) LN1: xn = LN(x)   (bf16 for GEMM A)
  layernorm_kernel<false, false><<<TOK / 8, 256, 0, stream>>>(x, nullptr, g1, be1, xn, nullptr);

  // 3) QKV projections
  dim3 ggrid(DD / 64, TOK / 128);
  gemm_wmma_kernel<0><<<ggrid, 256, 0, stream>>>(xn, wqT, nullptr, qb,   nullptr);
  gemm_wmma_kernel<0><<<ggrid, 256, 0, stream>>>(xn, wkT, nullptr, kbuf, nullptr);
  gemm_wmma_kernel<1><<<ggrid, 256, 0, stream>>>(xn, wvT, nullptr, vT,   nullptr);

  // 4) attention -> ctx (f32)
  attn_flash_kernel<<<(32 * HH * (SS / 16)) / 4, 128, 0, stream>>>(qb, kbuf, vT, ctx);

  // 5) LN2: a = LN(ctx + x); store a into d_out (f32) and ab (bf16)
  layernorm_kernel<true, true><<<TOK / 8, 256, 0, stream>>>(ctx, x, g2, be2, ab, out);

  // 6) FFN: hid = relu(a @ wf1 + bf1);  out += hid @ wf2 + bf2
  gemm_wmma_kernel<2><<<ggrid, 256, 0, stream>>>(ab,  wf1T, bf1, hid, nullptr);
  gemm_wmma_kernel<3><<<ggrid, 256, 0, stream>>>(hid, wf2T, bf2, nullptr, out);
}
